// TemporalAttentionPooling_34677565948183
// MI455X (gfx1250) — matrix operand order repaired and checked by reading, compiled, not run on hardware
//
#include <hip/hip_runtime.h>
#include <hip/hip_bf16.h>
#include <math.h>

typedef __attribute__((ext_vector_type(8)))  __bf16 v8bf;
typedef __attribute__((ext_vector_type(16))) __bf16 v16bf;
typedef __attribute__((ext_vector_type(8)))  float  v8f;
typedef __attribute__((ext_vector_type(4)))  int    v4i;

// address-space qualified int4 for the async-to-LDS builtin's pointer params
typedef __attribute__((address_space(1))) v4i gv4i;   // global
typedef __attribute__((address_space(3))) v4i lv4i;   // LDS

#define DIM   128
#define DIMP  132        // padded f32 row (528B stride -> conflict-free tiled loads)
#define DH    32
#define CHUNK 64

// ---- CDNA5 native transcendentals (guarded; fall back to libm) ----
#if __has_builtin(__builtin_amdgcn_tanhf)
#define FAST_TANH(x) __builtin_amdgcn_tanhf(x)
#else
#define FAST_TANH(x) tanhf(x)
#endif

#if __has_builtin(__builtin_amdgcn_exp2f)
#define FAST_EXP(x) __builtin_amdgcn_exp2f((x) * 1.4426950408889634f)
#else
#define FAST_EXP(x) __expf(x)
#endif

#if __has_builtin(__builtin_amdgcn_global_load_async_to_lds_b128) && \
    __has_builtin(__builtin_amdgcn_s_wait_asynccnt)
#define HAVE_ASYNC_LDS 1
#else
#define HAVE_ASYNC_LDS 0
#endif

__global__ __launch_bounds__(128)
void tap_kernel(const float* __restrict__ x, const int* __restrict__ ids,
                const float* __restrict__ W1, const float* __restrict__ b1,
                const float* __restrict__ W2, const float* __restrict__ b2,
                float* __restrict__ out, int N, int P)
{
    const int p    = blockIdx.x;
    const int tid  = threadIdx.x;
    const int lane = tid & 31;
    const int w    = tid >> 5;     // wave id (0..3), each owns a 16-row tile
    const int n    = lane & 15;    // N index within 16-wide tile
    const int half = lane >> 4;    // lane half selects K sub-range

    __shared__ __attribute__((aligned(16))) float  Xf[CHUNK][DIMP];     // 33 KB f32 tile
    __shared__ __attribute__((aligned(16))) __bf16 W1t[DH][DIM + 8];    // 8.5 KB W1^T bf16
    __shared__ float  sc[CHUNK];
    __shared__ float  ev[CHUNK];
    __shared__ int    seg[2];
    __shared__ float  s_runm, s_rund, s_newm, s_resc;

    // ---- segment bounds via binary search (ids sorted ascending) ----
    if (tid == 0) {
        int lo = 0, hi = N;
        while (lo < hi) { int m = (lo + hi) >> 1; if (ids[m] <  p) lo = m + 1; else hi = m; }
        seg[0] = lo;
        hi = N;
        while (lo < hi) { int m = (lo + hi) >> 1; if (ids[m] <= p) lo = m + 1; else hi = m; }
        seg[1] = lo;
        s_runm = -__builtin_inff();
        s_rund = 0.f;
    }
    // ---- stage W1 transposed as bf16 (padded rows -> conflict-free ds_load_b128) ----
    for (int i = tid; i < DIM * DH; i += 128) {
        int d = i >> 5, j = i & 31;            // W1 is (D=128, dh=32) row-major
        W1t[j][d] = (__bf16)W1[i];
    }
    __syncthreads();

    const int start = seg[0];
    const int len   = seg[1] - seg[0];

    // per-lane MLP constants (N = n for tile0, 16+n for tile1)
    const float b1a = b1[n],      b1b = b1[16 + n];
    const float w2a = W2[n],      w2b = W2[16 + n];
    const float b2v = b2[0];

    // ---- build B fragments once (reused for every chunk) ----
    // B layout (32x16, bf16): lane = half*16+n holds K = half*16 + e, N = n
    v16bf bf0[4], bf1[4];
#pragma unroll
    for (int kk = 0; kk < 4; ++kk) {
        const int kb = kk * 32 + half * 16;
        v8bf lo0 = *(const v8bf*)&W1t[n][kb];
        v8bf hi0 = *(const v8bf*)&W1t[n][kb + 8];
        v8bf lo1 = *(const v8bf*)&W1t[16 + n][kb];
        v8bf hi1 = *(const v8bf*)&W1t[16 + n][kb + 8];
        bf0[kk] = __builtin_shufflevector(lo0, hi0, 0,1,2,3,4,5,6,7,8,9,10,11,12,13,14,15);
        bf1[kk] = __builtin_shufflevector(lo1, hi1, 0,1,2,3,4,5,6,7,8,9,10,11,12,13,14,15);
    }

    float acc = 0.f;                       // thread tid owns pooled column tid
    const int nchunks = (len + CHUNK - 1) / CHUNK;

    for (int c = 0; c < nchunks; ++c) {
        const int lenc = min(CHUNK, len - c * CHUNK);
        const float* xb = x + (size_t)(start + c * CHUNK) * DIM;

        // ---- chunk copy global -> LDS (async-to-LDS when available) ----
        for (int i = tid; i < CHUNK * (DIM / 4); i += 128) {
            const int r  = i >> 5;
            const int c4 = (i & 31) * 4;
            if (r < lenc) {
#if HAVE_ASYNC_LDS
                __builtin_amdgcn_global_load_async_to_lds_b128(
                    (gv4i*)(xb + (size_t)r * DIM + c4),
                    (lv4i*)&Xf[r][c4], 0, 0);
#else
                *(float4*)&Xf[r][c4] = *(const float4*)(xb + (size_t)r * DIM + c4);
#endif
            } else {
                *(float4*)&Xf[r][c4] = make_float4(0.f, 0.f, 0.f, 0.f);
            }
        }
        if (c + 1 < nchunks) {
            // pull next 32KB chunk toward the caches (global_prefetch_b8)
            __builtin_prefetch((const char*)(xb + CHUNK * DIM) + tid * 256, 0, 1);
        }
#if HAVE_ASYNC_LDS
        __builtin_amdgcn_s_wait_asynccnt(0);
#endif
        __syncthreads();

        // ---- H = X @ W1 via v_wmma_f32_16x16x32_bf16, K=128 in 4 steps ----
        // A layout (16x32, bf16): lane holds M = lane&15; K = e + half*16
        // f32 tile is converted to bf16 in registers while building the fragment.
        v8f a0 = {}; v8f a1 = {};
        const int arow = (w << 4) + n;
#pragma unroll
        for (int kk = 0; kk < 4; ++kk) {
            const int kb = kk * 32 + half * 16;
            float4 f0 = *(const float4*)&Xf[arow][kb];
            float4 f1 = *(const float4*)&Xf[arow][kb + 4];
            float4 f2 = *(const float4*)&Xf[arow][kb + 8];
            float4 f3 = *(const float4*)&Xf[arow][kb + 12];
            v16bf a = { (__bf16)f0.x, (__bf16)f0.y, (__bf16)f0.z, (__bf16)f0.w,
                        (__bf16)f1.x, (__bf16)f1.y, (__bf16)f1.z, (__bf16)f1.w,
                        (__bf16)f2.x, (__bf16)f2.y, (__bf16)f2.z, (__bf16)f2.w,
                        (__bf16)f3.x, (__bf16)f3.y, (__bf16)f3.z, (__bf16)f3.w };
            a0 = __builtin_amdgcn_wmma_f32_16x16x32_bf16(false, a, false, bf0[kk],
                                                         (short)0, a0, false, false);
            a1 = __builtin_amdgcn_wmma_f32_16x16x32_bf16(false, a, false, bf1[kk],
                                                         (short)0, a1, false, false);
        }

        // ---- scores = tanh(H + b1) @ W2 + b2 ; reduce over dh across 16-lane halves ----
        // C layout: element e of lane (half,n) holds (M = e + half*8, N = n [+16 for tile1])
#pragma unroll
        for (int e = 0; e < 8; ++e) {
            float cd = FAST_TANH(a0[e] + b1a) * w2a + FAST_TANH(a1[e] + b1b) * w2b;
            cd += __shfl_xor(cd, 1);
            cd += __shfl_xor(cd, 2);
            cd += __shfl_xor(cd, 4);
            cd += __shfl_xor(cd, 8);
            if (n == 0) sc[(w << 4) + half * 8 + e] = cd + b2v;
        }
        __syncthreads();

        // ---- online softmax stats (wave 0) ----
        if (w == 0) {
            float m0 = (lane      < lenc) ? sc[lane]      : -__builtin_inff();
            float m1 = (lane + 32 < lenc) ? sc[lane + 32] : -__builtin_inff();
            float m = fmaxf(m0, m1);
            m = fmaxf(m, __shfl_xor(m, 1));
            m = fmaxf(m, __shfl_xor(m, 2));
            m = fmaxf(m, __shfl_xor(m, 4));
            m = fmaxf(m, __shfl_xor(m, 8));
            m = fmaxf(m, __shfl_xor(m, 16));
            if (lane == 0) {
                const float runm = s_runm;
                const float newm = fmaxf(runm, m);
                s_resc = (runm == -__builtin_inff()) ? 0.f : FAST_EXP(runm - newm);
                s_newm = newm;
                s_runm = newm;
            }
        }
        __syncthreads();

        if (tid < CHUNK)
            ev[tid] = (tid < lenc) ? FAST_EXP(sc[tid] - s_newm) : 0.f;
        __syncthreads();

        if (w == 0) {
            float s = ev[lane] + ev[lane + 32];
            s += __shfl_xor(s, 1);
            s += __shfl_xor(s, 2);
            s += __shfl_xor(s, 4);
            s += __shfl_xor(s, 8);
            s += __shfl_xor(s, 16);
            if (lane == 0) s_rund = s_rund * s_resc + s;
        }

        // ---- weighted accumulate: column tid, exact f32 rows from LDS ----
        float a2 = acc * s_resc;
        for (int r = 0; r < lenc; ++r)
            a2 = fmaf(ev[r], Xf[r][tid], a2);
        acc = a2;
        __syncthreads();   // also protects LDS reuse next chunk & publishes s_rund
    }

    const float d = s_rund;
    out[(size_t)p * DIM + tid] = (d > 0.f) ? acc / d : 0.f;
    if (tid == 0) {
        int* pids = (int*)(out + (size_t)P * DIM);
        pids[p] = p;
    }
}

extern "C" void kernel_launch(void* const* d_in, const int* in_sizes, int n_in,
                              void* d_out, int out_size, void* d_ws, size_t ws_size,
                              hipStream_t stream) {
    const float* x   = (const float*)d_in[0];
    const int*   ids = (const int*)  d_in[1];
    const float* W1  = (const float*)d_in[2];
    const float* b1  = (const float*)d_in[3];
    const float* W2  = (const float*)d_in[4];
    const float* b2  = (const float*)d_in[5];
    float* out = (float*)d_out;

    const int N = in_sizes[1];              // one id per row
    const int P = out_size / (DIM + 1);     // pooled (P*128 f32) + pids (P i32)

    tap_kernel<<<P, 128, 0, stream>>>(x, ids, W1, b1, W2, b2, out, N, P);
}
